// GATBranch_20332375179335
// MI455X (gfx1250) — matrix operand order repaired and checked by reading, not compiled, not run on hardware
//
#include <hip/hip_runtime.h>
#include <hip/hip_bf16.h>

#define HID 128
#define HEADS 4
#define EF 4

typedef __attribute__((ext_vector_type(16))) __bf16 v16bf;
typedef __attribute__((ext_vector_type(8)))  __bf16 v8bf;
typedef __attribute__((ext_vector_type(8)))  float  v8f;

__device__ __forceinline__ __bf16 f2bf(float f) {
  unsigned u = __float_as_uint(f);
  unsigned r = (u + 0x7FFFu + ((u >> 16) & 1u)) >> 16;   // round-to-nearest-even
  unsigned short s = (unsigned short)r;
  __bf16 b;
  __builtin_memcpy(&b, &s, 2);
  return b;
}

// Monotone float<->uint mapping so atomicMax(uint) == float max. 0 == "-inf".
__device__ __forceinline__ unsigned f2ord(float f) {
  unsigned u = __float_as_uint(f);
  return (u & 0x80000000u) ? ~u : (u | 0x80000000u);
}
__device__ __forceinline__ float ord2f(unsigned u) {
  unsigned b = (u & 0x80000000u) ? (u & 0x7FFFFFFFu) : ~u;
  return __uint_as_float(b);
}

// ---------------------------------------------------------------------------
// WMMA GEMM: C[n x m] = A[n x k](bf16) * BT[m x k](bf16)^T (+ bias), f32 out.
// One wave per 16x16 tile; K in chunks of 32 via v_wmma_f32_16x16x32_bf16.
// n multiple of 16, k multiple of 32, m multiple of 16 (holds for all calls).
// ---------------------------------------------------------------------------
__global__ void gemm_bf16_wmma(const __bf16* __restrict__ A,
                               const __bf16* __restrict__ BT,
                               const float* __restrict__ bias,
                               float* __restrict__ C,
                               int n, int k, int m)
{
  int wave = (int)((blockIdx.x * blockDim.x + threadIdx.x) >> 5);
  int lane = threadIdx.x & 31;
  int tilesM = m >> 4;
  int tilesN = n >> 4;
  if (wave >= tilesN * tilesM) return;          // wave-uniform guard
  int tn = wave / tilesM;
  int tm = wave % tilesM;
  int half = lane >> 4;                          // 0 or 1
  int lrow = lane & 15;
  // A lane layout (16x32 bf16): row = lane&15, K = half*8 + {0..7, 16..23}
  const __bf16* arow = A + (size_t)(tn * 16 + lrow) * k + half * 8;
  // B lane layout (32x16 bf16): col = lane&15, K = half*8 + {0..7, 16..23}
  const __bf16* brow = BT + (size_t)(tm * 16 + lrow) * k + half * 8;
  v8f acc = {};
  for (int k0 = 0; k0 < k; k0 += 32) {
    union { v16bf v; v8bf h[2]; } ua, ub;
    ua.h[0] = *(const v8bf*)(arow + k0);
    ua.h[1] = *(const v8bf*)(arow + k0 + 16);
    ub.h[0] = *(const v8bf*)(brow + k0);
    ub.h[1] = *(const v8bf*)(brow + k0 + 16);
    acc = __builtin_amdgcn_wmma_f32_16x16x32_bf16(false, ua.v, false, ub.v,
                                                  (short)0, acc, false, false);
  }
  int col = tm * 16 + lrow;
  float bc = bias ? bias[col] : 0.0f;
  int rbase = tn * 16 + half * 8;                // C: VGPR v -> row v + 8*half
#pragma unroll
  for (int v = 0; v < 8; ++v)
    C[(size_t)(rbase + v) * m + col] = acc[v] + bc;
}

// ---------------------------------------------------------------------------
// Graph / elementwise kernels
// ---------------------------------------------------------------------------
__global__ void loopsum_kernel(const int* __restrict__ dst, const float* __restrict__ eattr,
                               float* __restrict__ loop_sum, float* __restrict__ deg, int E)
{
  int e = blockIdx.x * blockDim.x + threadIdx.x;
  if (e >= E) return;
  int d = dst[e];
#pragma unroll
  for (int f = 0; f < EF; ++f)
    atomicAdd(&loop_sum[(size_t)d * EF + f], eattr[(size_t)e * EF + f]);
  atomicAdd(&deg[d], 1.0f);
}

__global__ void build_ea_kernel(const float* __restrict__ eattr, const float* __restrict__ loop_sum,
                                const float* __restrict__ deg, float* __restrict__ ea,
                                int Ereal, int n)
{
  long idx = (long)blockIdx.x * blockDim.x + threadIdx.x;
  if (idx >= (long)(Ereal + n) * EF) return;
  int e = (int)(idx / EF), f = (int)(idx % EF);
  if (e < Ereal) ea[idx] = eattr[idx];
  else {
    int node = e - Ereal;
    ea[idx] = loop_sum[(size_t)node * EF + f] / fmaxf(deg[node], 1.0f);
  }
}

__global__ void proj_kernel(const float* __restrict__ x, const float* __restrict__ Wp,
                            const float* __restrict__ bp, __bf16* __restrict__ h, int n)
{
  long idx = (long)blockIdx.x * blockDim.x + threadIdx.x;
  if (idx >= (long)n * HID) return;
  int node = (int)(idx >> 7), k = (int)(idx & 127);
  float s = bp[k];
#pragma unroll
  for (int f = 0; f < 4; ++f) s += x[(size_t)node * 4 + f] * Wp[f * HID + k];
  h[idx] = f2bf(s);
}

__global__ void transpose_bf16_kernel(const float* __restrict__ W, __bf16* __restrict__ WT,
                                      int K, int M)
{
  long idx = (long)blockIdx.x * blockDim.x + threadIdx.x;
  if (idx >= (long)K * M) return;
  int kk = (int)(idx / M), mm = (int)(idx % M);
  WT[(size_t)mm * K + kk] = f2bf(W[idx]);
}

// weA[f][h] = sum_c We[f, h*C+c] * ae[h, c]   (folds eh away entirely)
__global__ void wealpha_kernel(const float* __restrict__ We, const float* __restrict__ ae,
                               float* __restrict__ out, int ef, int heads, int C)
{
  int idx = blockIdx.x * blockDim.x + threadIdx.x;
  if (idx >= ef * heads) return;
  int f = idx / heads, h = idx % heads;
  float s = 0.f;
  for (int c = 0; c < C; ++c) s += We[(size_t)f * heads * C + h * C + c] * ae[h * C + c];
  out[f * heads + h] = s;
}

__global__ void alpha_sd_kernel(const float* __restrict__ xh, const float* __restrict__ as_,
                                const float* __restrict__ ad_, float* __restrict__ als,
                                float* __restrict__ ald, int n, int heads, int C)
{
  long idx = (long)blockIdx.x * blockDim.x + threadIdx.x;
  if (idx >= (long)n * heads) return;
  int node = (int)(idx / heads);
  int h = (int)(idx % heads);
  const float* row = xh + (size_t)node * heads * C + (size_t)h * C;
  float s = 0.f, d = 0.f;
  for (int c = 0; c < C; ++c) { float v = row[c]; s += v * as_[h * C + c]; d += v * ad_[h * C + c]; }
  als[idx] = s; ald[idx] = d;
}

__global__ void logits_kernel(const float* __restrict__ ea, const int* __restrict__ src,
                              const int* __restrict__ dst, const float* __restrict__ als,
                              const float* __restrict__ ald, const float* __restrict__ weA,
                              float* __restrict__ logits, unsigned* __restrict__ mx,
                              int Etot, int Ereal, int heads)
{
  long idx = (long)blockIdx.x * blockDim.x + threadIdx.x;
  if (idx >= (long)Etot * heads) return;
  int e = (int)(idx / heads);
  int h = (int)(idx % heads);
  int s, d;
  if (e < Ereal) { s = src[e]; d = dst[e]; } else { s = d = e - Ereal; }
  float ale = 0.f;
#pragma unroll
  for (int f = 0; f < EF; ++f) ale += ea[(size_t)e * EF + f] * weA[f * heads + h];
  float v = als[(size_t)s * heads + h] + ald[(size_t)d * heads + h] + ale;
  v = v > 0.f ? v : 0.2f * v;                    // leaky_relu(0.2)
  logits[idx] = v;
  atomicMax(&mx[(size_t)d * heads + h], f2ord(v));
}

__global__ void expsum_kernel(const int* __restrict__ dst, float* __restrict__ ev,
                              const unsigned* __restrict__ mx, float* __restrict__ den,
                              int Etot, int Ereal, int heads)
{
  long idx = (long)blockIdx.x * blockDim.x + threadIdx.x;
  if (idx >= (long)Etot * heads) return;
  int e = (int)(idx / heads);
  int h = (int)(idx % heads);
  int d = (e < Ereal) ? dst[e] : e - Ereal;
  float m = ord2f(mx[(size_t)d * heads + h]);
  float ex = expf(ev[idx] - m);
  ev[idx] = ex;                                   // in-place logits -> exp
  atomicAdd(&den[(size_t)d * heads + h], ex);
}

__global__ void normalize_kernel(const int* __restrict__ dst, float* __restrict__ ev,
                                 const float* __restrict__ den, int Etot, int Ereal, int heads)
{
  long idx = (long)blockIdx.x * blockDim.x + threadIdx.x;
  if (idx >= (long)Etot * heads) return;
  int e = (int)(idx / heads);
  int h = (int)(idx % heads);
  int d = (e < Ereal) ? dst[e] : e - Ereal;
  ev[idx] /= den[(size_t)d * heads + h];
}

__global__ void aggregate_kernel(const int* __restrict__ src, const int* __restrict__ dst,
                                 const float* __restrict__ xh, const float* __restrict__ alpha,
                                 float* __restrict__ out, int Etot, int Ereal, int heads, int HC)
{
  long idx = (long)blockIdx.x * blockDim.x + threadIdx.x;
  if (idx >= (long)Etot * HC) return;
  int e = (int)(idx / HC);
  int j = (int)(idx % HC);
  int C = HC / heads;
  int h = j / C;
  int s, d;
  if (e < Ereal) { s = src[e]; d = dst[e]; } else { s = d = e - Ereal; }
  atomicAdd(&out[(size_t)d * HC + j], xh[(size_t)s * HC + j] * alpha[(size_t)e * heads + h]);
}

// LayerNorm(in + bias)*g + be, optional ELU, write bf16. One wave32 per node.
__global__ void ln_act_bf16_kernel(const float* __restrict__ in, const float* __restrict__ bias,
                                   const float* __restrict__ g, const float* __restrict__ be,
                                   __bf16* __restrict__ out, int n, int c, int do_elu)
{
  int wave = (int)((blockIdx.x * blockDim.x + threadIdx.x) >> 5);
  int lane = threadIdx.x & 31;
  if (wave >= n) return;
  int per = c >> 5;                               // 16 for c=512, 4 for c=128
  const float* row = in + (size_t)wave * c;
  float vals[16];
  float s = 0.f;
  for (int i = 0; i < per; ++i) {
    int j = lane * per + i;
    vals[i] = row[j] + bias[j];
    s += vals[i];
  }
  for (int o = 16; o >= 1; o >>= 1) s += __shfl_xor(s, o, 32);
  float mu = s / c;
  float vs = 0.f;
  for (int i = 0; i < per; ++i) { float d = vals[i] - mu; vs += d * d; }
  for (int o = 16; o >= 1; o >>= 1) vs += __shfl_xor(vs, o, 32);
  float inv = rsqrtf(vs / c + 1e-5f);
  for (int i = 0; i < per; ++i) {
    int j = lane * per + i;
    float y = (vals[i] - mu) * inv * g[j] + be[j];
    if (do_elu) y = (y > 0.f) ? y : (expf(y) - 1.f);
    out[(size_t)wave * c + j] = f2bf(y);
  }
}

__global__ void importance_kernel(const float* __restrict__ emb, const float* __restrict__ Wi,
                                  const float* __restrict__ bi, float* __restrict__ out, int n)
{
  int i = blockIdx.x * blockDim.x + threadIdx.x;
  if (i >= n) return;
  float s = bi[0];
  const float* row = emb + (size_t)i * HID;
  for (int k = 0; k < HID; ++k) s += row[k] * Wi[k];
  out[i] = 1.f / (1.f + expf(-s));
}

// ---------------------------------------------------------------------------
extern "C" void kernel_launch(void* const* d_in, const int* in_sizes, int n_in,
                              void* d_out, int out_size, void* d_ws, size_t ws_size,
                              hipStream_t stream)
{
  (void)n_in; (void)out_size; (void)ws_size;
  const float* x    = (const float*)d_in[0];
  const int*   eidx = (const int*)d_in[1];
  const float* eattr= (const float*)d_in[2];
  const float* Wp   = (const float*)d_in[3];
  const float* bp   = (const float*)d_in[4];
  const float* W1   = (const float*)d_in[5];
  const float* We1  = (const float*)d_in[6];
  const float* as1  = (const float*)d_in[7];
  const float* ad1  = (const float*)d_in[8];
  const float* ae1  = (const float*)d_in[9];
  const float* b1   = (const float*)d_in[10];
  const float* g1   = (const float*)d_in[11];
  const float* be1  = (const float*)d_in[12];
  const float* W2   = (const float*)d_in[13];
  const float* We2  = (const float*)d_in[14];
  const float* as2  = (const float*)d_in[15];
  const float* ad2  = (const float*)d_in[16];
  const float* ae2  = (const float*)d_in[17];
  const float* b2   = (const float*)d_in[18];
  const float* g2   = (const float*)d_in[19];
  const float* be2  = (const float*)d_in[20];
  const float* Wo   = (const float*)d_in[21];
  const float* bo   = (const float*)d_in[22];
  const float* Wi   = (const float*)d_in[23];
  const float* bi   = (const float*)d_in[24];

  const int n    = in_sizes[0] / 4;     // 20000 (multiple of 16)
  const int E    = in_sizes[2] / 4;     // 320000
  const int Etot = E + n;
  const int* srcI = eidx;
  const int* dstI = eidx + E;
  const int HC1 = HEADS * HID;          // 512

  // ---- scratch layout (bump allocator, 256B aligned) ----
  size_t off = 0;
  auto alloc = [&](size_t bytes) -> void* {
    void* p = (char*)d_ws + off;
    off += bytes; off = (off + 255) & ~(size_t)255;
    return p;
  };
  auto a256 = [](size_t b) { return (b + 255) & ~(size_t)255; };

  float*    loop_sum = (float*)alloc((size_t)n * EF * 4);
  float*    deg      = (float*)alloc((size_t)n * 4);
  float*    ea       = (float*)alloc((size_t)Etot * EF * 4);
  __bf16*   h_bf     = (__bf16*)alloc((size_t)n * HID * 2);
  __bf16*   W1T      = (__bf16*)alloc((size_t)HC1 * HID * 2);     // [512][128]
  __bf16*   W2T      = (__bf16*)alloc((size_t)HID * HC1 * 2);     // [128][512]
  __bf16*   WoT      = (__bf16*)alloc((size_t)HID * HID * 2);     // [128][128]
  float*    weA1     = (float*)alloc(EF * HEADS * 4);
  float*    weA2     = (float*)alloc(EF * 4);
  float*    als1     = (float*)alloc((size_t)n * HEADS * 4);
  float*    ald1     = (float*)alloc((size_t)n * HEADS * 4);
  unsigned* max1     = (unsigned*)alloc((size_t)n * HEADS * 4);
  float*    den1     = (float*)alloc((size_t)n * HEADS * 4);
  float*    alpha1   = (float*)alloc((size_t)Etot * HEADS * 4);
  float*    als2     = (float*)alloc((size_t)n * 4);
  float*    ald2     = (float*)alloc((size_t)n * 4);
  unsigned* max2     = (unsigned*)alloc((size_t)n * 4);
  float*    den2     = (float*)alloc((size_t)n * 4);
  float*    out1     = (float*)alloc((size_t)n * HC1 * 4);

  // region1: xh1 (f32 NxHC1) first; reused (after layer-1 aggregation) for
  // h1n(bf16 NxHC1) + xh2(f32 NxHID) + out2(f32 NxHID) + h2n(bf16 NxHID)
  size_t sz_xh1 = (size_t)n * HC1 * 4;
  size_t o_xh2  = a256((size_t)n * HC1 * 2);
  size_t o_out2 = o_xh2 + a256((size_t)n * HID * 4);
  size_t o_h2n  = o_out2 + a256((size_t)n * HID * 4);
  size_t sz_r1b = o_h2n + (size_t)n * HID * 2;
  char* region1 = (char*)alloc(sz_xh1 > sz_r1b ? sz_xh1 : sz_r1b);
  float*  xh1 = (float*)region1;
  __bf16* h1n = (__bf16*)region1;
  float*  xh2 = (float*)(region1 + o_xh2);
  float*  out2= (float*)(region1 + o_out2);
  __bf16* h2n = (__bf16*)(region1 + o_h2n);

  float* outp = (float*)d_out;
  float* att  = outp + (size_t)n * HID;   // edge_attention region [Etot]
  float* imp  = att + Etot;               // importance region [n]

  const int T = 256;
  auto blocks = [](long total, int t) { return (int)((total + t - 1) / t); };

  // ---- zero accumulators (graph-capture-safe stream memsets) ----
  (void)hipMemsetAsync(loop_sum, 0, (size_t)n * EF * 4, stream);
  (void)hipMemsetAsync(deg,      0, (size_t)n * 4, stream);
  (void)hipMemsetAsync(max1,     0, (size_t)n * HEADS * 4, stream);
  (void)hipMemsetAsync(den1,     0, (size_t)n * HEADS * 4, stream);
  (void)hipMemsetAsync(max2,     0, (size_t)n * 4, stream);
  (void)hipMemsetAsync(den2,     0, (size_t)n * 4, stream);
  (void)hipMemsetAsync(out1,     0, (size_t)n * HC1 * 4, stream);

  // ---- self-loop attrs + concatenated edge features ----
  loopsum_kernel<<<blocks(E, T), T, 0, stream>>>(dstI, eattr, loop_sum, deg, E);
  build_ea_kernel<<<blocks((long)Etot * EF, T), T, 0, stream>>>(eattr, loop_sum, deg, ea, E, n);

  // ---- input projection + weight prep (bf16) ----
  proj_kernel<<<blocks((long)n * HID, T), T, 0, stream>>>(x, Wp, bp, h_bf, n);
  transpose_bf16_kernel<<<blocks((long)HID * HC1, T), T, 0, stream>>>(W1, W1T, HID, HC1);
  transpose_bf16_kernel<<<blocks((long)HC1 * HID, T), T, 0, stream>>>(W2, W2T, HC1, HID);
  transpose_bf16_kernel<<<blocks((long)HID * HID, T), T, 0, stream>>>(Wo, WoT, HID, HID);
  wealpha_kernel<<<1, 64, 0, stream>>>(We1, ae1, weA1, EF, HEADS, HID);
  wealpha_kernel<<<1, 64, 0, stream>>>(We2, ae2, weA2, EF, 1, HID);

  // ---- GAT layer 1 ----
  {
    long waves = (long)(n / 16) * (HC1 / 16);
    gemm_bf16_wmma<<<blocks(waves * 32, T), T, 0, stream>>>(h_bf, W1T, nullptr, xh1, n, HID, HC1);
  }
  alpha_sd_kernel<<<blocks((long)n * HEADS, T), T, 0, stream>>>(xh1, as1, ad1, als1, ald1, n, HEADS, HID);
  logits_kernel<<<blocks((long)Etot * HEADS, T), T, 0, stream>>>(ea, srcI, dstI, als1, ald1, weA1,
                                                                 alpha1, max1, Etot, E, HEADS);
  expsum_kernel<<<blocks((long)Etot * HEADS, T), T, 0, stream>>>(dstI, alpha1, max1, den1, Etot, E, HEADS);
  normalize_kernel<<<blocks((long)Etot * HEADS, T), T, 0, stream>>>(dstI, alpha1, den1, Etot, E, HEADS);
  aggregate_kernel<<<blocks((long)Etot * HC1, T), T, 0, stream>>>(srcI, dstI, xh1, alpha1, out1,
                                                                  Etot, E, HEADS, HC1);
  ln_act_bf16_kernel<<<blocks((long)n * 32, T), T, 0, stream>>>(out1, b1, g1, be1, h1n, n, HC1, 1);
  // xh1 now dead; region1 reused.

  // ---- GAT layer 2 (1 head, mean == identity) ----
  (void)hipMemsetAsync(out2, 0, (size_t)n * HID * 4, stream);
  {
    long waves = (long)(n / 16) * (HID / 16);
    gemm_bf16_wmma<<<blocks(waves * 32, T), T, 0, stream>>>(h1n, W2T, nullptr, xh2, n, HC1, HID);
  }
  alpha_sd_kernel<<<blocks((long)n, T), T, 0, stream>>>(xh2, as2, ad2, als2, ald2, n, 1, HID);
  logits_kernel<<<blocks((long)Etot, T), T, 0, stream>>>(ea, srcI, dstI, als2, ald2, weA2,
                                                         att, max2, Etot, E, 1);
  expsum_kernel<<<blocks((long)Etot, T), T, 0, stream>>>(dstI, att, max2, den2, Etot, E, 1);
  normalize_kernel<<<blocks((long)Etot, T), T, 0, stream>>>(dstI, att, den2, Etot, E, 1); // final alpha out
  aggregate_kernel<<<blocks((long)Etot * HID, T), T, 0, stream>>>(srcI, dstI, xh2, att, out2,
                                                                  Etot, E, 1, HID);
  ln_act_bf16_kernel<<<blocks((long)n * 32, T), T, 0, stream>>>(out2, b2, g2, be2, h2n, n, HID, 0);

  // ---- output projection + importance ----
  {
    long waves = (long)(n / 16) * (HID / 16);
    gemm_bf16_wmma<<<blocks(waves * 32, T), T, 0, stream>>>(h2n, WoT, bo, outp, n, HID, HID);
  }
  importance_kernel<<<blocks((long)n, T), T, 0, stream>>>(outp, Wi, bi, imp, n);
}
